// DyGANGenerator_16509854285885
// MI455X (gfx1250) — matrix operand order repaired and checked by reading, not compiled, not run on hardware
//
#include <hip/hip_runtime.h>

#define B_   256
#define ZD_  128
#define HIN_ 128
#define H_   512
#define HT_  32
#define N_   16384
#define RW_  16
#define KSPLIT_ 32

typedef __attribute__((ext_vector_type(16))) __bf16        v16bf;
typedef __attribute__((ext_vector_type(8)))  float         v8f;
typedef __attribute__((ext_vector_type(4)))  unsigned int  u32x4;

__device__ __forceinline__ unsigned short f2bf(float f) {
    union { float f; unsigned int u; } x; x.f = f;
    unsigned int u = x.u;
    u += 0x7FFFu + ((u >> 16) & 1u);          // round-to-nearest-even
    return (unsigned short)(u >> 16);
}
__device__ __forceinline__ float sigf(float x) { return 1.0f / (1.0f + __expf(-x)); }

// ---------------------------------------------------------------------------
// bf16 WMMA GEMM: C[M,N] = A[M,K] * B[N,K]^T + bias[N]
// A, B row-major bf16 (K contiguous).
// 8 waves/block; each wave computes a 16(M) x 64(N) tile.
//   WM = waves along M (template): WM=2 -> block tile 32x256 (wide-N GEMMs)
//                                  WM=4 -> block tile 64x128 (narrow-N GEMM)
// Split-K: gridDim.z = #K-chunks; each block reduces kc = K/gridDim.z and
// writes its partial to C + z*M*N (bias must be null when gridDim.z > 1).
// Requirements (all call sites satisfy): kc % 64 == 0, M/N tile-aligned.
// Wave id is scalarized so all guards are scalar branches -> EXEC stays
// all-ones around every v_wmma (ISA 7.12). K loop is unrolled x2 with
// double-buffered fragments so global_load_b128 overlaps v_wmma.
// ---------------------------------------------------------------------------
struct FragPair { u32x4 q0, q1; };

__device__ __forceinline__ void ldA(FragPair& f, const unsigned short* p) {
    f.q0 = *(const u32x4*)(p);          // K = hh*8 .. +7
    f.q1 = *(const u32x4*)(p + 16);     // K = 16+hh*8 .. +7
}
__device__ __forceinline__ void ldB(FragPair& f, const unsigned short* p) {
    f.q0 = *(const u32x4*)(p);          // K = hh*8 .. +7
    f.q1 = *(const u32x4*)(p + 16);     // K = 16+hh*8 .. +7
}
__device__ __forceinline__ v8f mma(const FragPair& a, const FragPair& b, v8f c) {
    union { v16bf v; u32x4 q[2]; } av, bv;
    av.q[0] = a.q0; av.q[1] = a.q1;
    bv.q[0] = b.q0; bv.q[1] = b.q1;
    return __builtin_amdgcn_wmma_f32_16x16x32_bf16(false, av.v, false, bv.v,
                                                   (short)0, c, false, false);
}

template<int WM>
__global__ __launch_bounds__(256)
void gemm_bf16_wmma_k(const unsigned short* __restrict__ A,
                      const unsigned short* __restrict__ Bm,
                      const float* __restrict__ bias,
                      float* __restrict__ C,
                      int M, int N, int K)
{
    const int lane = threadIdx.x & 31;
    const int wave = __builtin_amdgcn_readfirstlane(threadIdx.x >> 5);
    const int wm   = wave % WM;
    const int wn   = wave / WM;
    const int m0    = blockIdx.x * (16 * WM) + wm * 16;
    const int nbase = blockIdx.y * (64 * (8 / WM)) + wn * 64;
    if (m0 >= M || nbase + 64 > N) return;     // scalar branch (uniform)

    const int kc   = K / gridDim.z;            // K-chunk for this z-slice
    const int kbeg = blockIdx.z * kc;

    const int r16 = lane & 15;
    const int hh  = lane >> 4;

    const unsigned short* arow = A + (size_t)(m0 + r16) * K + kbeg + hh * 8;
    const unsigned short* brow = Bm + (size_t)(nbase + r16) * K + kbeg + hh * 8;
    const size_t bTile = (size_t)16 * K;

    v8f acc0 = {}, acc1 = {}, acc2 = {}, acc3 = {};
    FragPair a0, a1, b0[4], b1[4];

    ldA(a0, arow);
#pragma unroll
    for (int i = 0; i < 4; ++i) ldB(b0[i], brow + i * bTile);

    for (int k0 = 0; k0 < kc; k0 += 64) {
        // load the k+32 tile set while computing the k tile
        ldA(a1, arow + k0 + 32);
#pragma unroll
        for (int i = 0; i < 4; ++i) ldB(b1[i], brow + k0 + 32 + i * bTile);
        __builtin_prefetch((const void*)(arow + k0 + 128), 0, 3);

        acc0 = mma(a0, b0[0], acc0);
        acc1 = mma(a0, b0[1], acc1);
        acc2 = mma(a0, b0[2], acc2);
        acc3 = mma(a0, b0[3], acc3);

        if (k0 + 64 < kc) {                     // scalar branch
            ldA(a0, arow + k0 + 64);
#pragma unroll
            for (int i = 0; i < 4; ++i) ldB(b0[i], brow + k0 + 64 + i * bTile);
        }

        acc0 = mma(a1, b1[0], acc0);
        acc1 = mma(a1, b1[1], acc1);
        acc2 = mma(a1, b1[2], acc2);
        acc3 = mma(a1, b1[3], acc3);
    }

    // C/D layout: lane -> col n = r16; VGPR r -> row m0 + hh*8 + r
    float* Cz = C + (size_t)blockIdx.z * ((size_t)M * N);
#define STORE_TILE(ACC, I)                                                     \
    {                                                                          \
        int col = nbase + (I) * 16 + r16;                                      \
        float bv = bias ? bias[col] : 0.0f;                                    \
        _Pragma("unroll")                                                      \
        for (int r = 0; r < 8; ++r)                                            \
            Cz[(size_t)(m0 + hh * 8 + r) * N + col] = ACC[r] + bv;             \
    }
    STORE_TILE(acc0, 0)
    STORE_TILE(acc1, 1)
    STORE_TILE(acc2, 2)
    STORE_TILE(acc3, 3)
#undef STORE_TILE
}

__global__ __launch_bounds__(256)
void reduce_splitk_k(const float* __restrict__ part, float* __restrict__ C,
                     int total, int S) {
    int i = blockIdx.x * blockDim.x + threadIdx.x;
    if (i >= total) return;
    float acc = 0.f;
    for (int s = 0; s < S; ++s) acc += part[(size_t)s * total + i];
    C[i] = acc;
}

// ---------------------------------------------------------------------------
// Small helper kernels (one-time conversions + elementwise stages)
// ---------------------------------------------------------------------------
__global__ void cvt_bf16_k(const float* __restrict__ in, unsigned short* __restrict__ out, size_t n) {
    size_t i = (size_t)blockIdx.x * blockDim.x + threadIdx.x;
    size_t st = (size_t)gridDim.x * blockDim.x;
    for (; i < n; i += st) out[i] = f2bf(in[i]);
}

__global__ void concat_wcat_k(const float* __restrict__ wih, const float* __restrict__ whh,
                              unsigned short* __restrict__ out) {
    const size_t n = (size_t)4 * H_ * (HIN_ + H_);
    size_t i = (size_t)blockIdx.x * blockDim.x + threadIdx.x;
    size_t st = (size_t)gridDim.x * blockDim.x;
    for (; i < n; i += st) {
        size_t r = i / (HIN_ + H_), c = i % (HIN_ + H_);
        float v = (c < HIN_) ? wih[r * HIN_ + c] : whh[r * H_ + (c - HIN_)];
        out[i] = f2bf(v);
    }
}

__global__ void concat_xh_k(const float* __restrict__ x, const float* __restrict__ h,
                            unsigned short* __restrict__ out) {
    const size_t n = (size_t)B_ * (HIN_ + H_);
    size_t i = (size_t)blockIdx.x * blockDim.x + threadIdx.x;
    size_t st = (size_t)gridDim.x * blockDim.x;
    for (; i < n; i += st) {
        size_t b = i / (HIN_ + H_), c = i % (HIN_ + H_);
        float v = (c < HIN_) ? x[b * HIN_ + c] : h[b * H_ + (c - HIN_)];
        out[i] = f2bf(v);
    }
}

__global__ void addvec_k(const float* a, const float* b, float* o, int n) {
    int i = blockIdx.x * blockDim.x + threadIdx.x;
    if (i < n) o[i] = a[i] + b[i];
}

__global__ void zero_k(float* p, int n) {
    int i = blockIdx.x * blockDim.x + threadIdx.x;
    if (i < n) p[i] = 0.0f;
}

__global__ __launch_bounds__(256)
void rowdot_tanh_k(const float* __restrict__ A, const float* __restrict__ W,
                   const float* __restrict__ bias, float* __restrict__ C, int N, int K) {
    int b = blockIdx.x;
    int j = blockIdx.y * blockDim.x + threadIdx.x;
    if (j >= N) return;
    const float* a = A + (size_t)b * K;
    const float* w = W + (size_t)j * K;
    float acc = bias[j];
    for (int k = 0; k < K; ++k) acc = fmaf(a[k], w[k], acc);
    C[(size_t)b * N + j] = tanhf(acc);
}

__global__ __launch_bounds__(256)
void lstm_elem_k(const float* __restrict__ gates, float* __restrict__ c,
                 float* __restrict__ h, unsigned short* __restrict__ hbf) {
    int i = blockIdx.x * 256 + threadIdx.x;      // B_*H_ threads
    int b = i >> 9, j = i & (H_ - 1);
    const float* g = gates + (size_t)b * 4 * H_;
    float ig = sigf(g[j]);
    float fg = sigf(g[H_ + j]);
    float gg = tanhf(g[2 * H_ + j]);
    float og = sigf(g[3 * H_ + j]);
    float c2 = fg * c[i] + ig * gg;
    float h2 = og * tanhf(c2);
    c[i] = c2; h[i] = h2; hbf[i] = f2bf(h2);
}

__global__ __launch_bounds__(256)
void softmax_g_k(const float* __restrict__ p, const float* __restrict__ gum,
                 float* __restrict__ rw, unsigned short* __restrict__ vbf) {
    __shared__ float red[256];
    int b = blockIdx.x, tid = threadIdx.x;
    const float* pr = p   + (size_t)b * N_;
    const float* gr = gum + (size_t)b * N_;
    float mx = -3.4e38f;
    for (int n = tid; n < N_; n += 256) mx = fmaxf(mx, pr[n] + gr[n]);
    red[tid] = mx; __syncthreads();
    for (int s = 128; s > 0; s >>= 1) { if (tid < s) red[tid] = fmaxf(red[tid], red[tid + s]); __syncthreads(); }
    mx = red[0]; __syncthreads();

    float sum = 0.f;
    float* outr = rw + (size_t)b * ((size_t)RW_ * N_);
    for (int n = tid; n < N_; n += 256) { float e = __expf(pr[n] + gr[n] - mx); outr[n] = e; sum += e; }
    red[tid] = sum; __syncthreads();
    for (int s = 128; s > 0; s >>= 1) { if (tid < s) red[tid] += red[tid + s]; __syncthreads(); }
    float inv = 1.0f / red[0];

    unsigned short* vb = vbf + (size_t)b * N_;
    for (int n = tid; n < N_; n += 256) { float v = outr[n] * inv; outr[n] = v; vb[n] = f2bf(v); }
}

__global__ void timedec_k(const float* __restrict__ h, const float* __restrict__ Wtd,
                          const float* __restrict__ btd, const float* __restrict__ Wtp,
                          const float* __restrict__ du, float* __restrict__ tpre) {
    __shared__ float red[HT_];
    int b = blockIdx.x, ht = threadIdx.x;        // 32 threads
    const float* hr = h + (size_t)b * H_;
    const float* w = Wtd + (size_t)ht * H_;
    float acc = btd[ht];
    for (int k = 0; k < H_; ++k) acc = fmaf(hr[k], w[k], acc);
    float td = tanhf(acc);
    float u = du[(size_t)b * HT_ + ht];
    td = (u < 0.8f) ? td * 1.25f : 0.0f;         // dropout p=0.2, train mode
    red[ht] = td * Wtp[ht];
    __syncthreads();
    if (ht == 0) {
        float s = 0.f;
        for (int k = 0; k < HT_; ++k) s += red[k];
        tpre[b] = fminf(s, 1.0f);
    }
}

__global__ __launch_bounds__(256)
void constraints_k(const float* __restrict__ tpre, float* __restrict__ lastt,
                   float* __restrict__ ts, float* __restrict__ Ht,
                   const float* __restrict__ Wtu, const float* __restrict__ btu, int step) {
    __shared__ float red[256];
    int b = threadIdx.x;
    float t = tpre[b];
    red[b] = t; __syncthreads();
    for (int s = 128; s > 0; s >>= 1) { if (b < s) red[b] = fminf(red[b], red[b + s]); __syncthreads(); }
    float mn = red[0]; __syncthreads();
    if (mn < 0.1f) t = t - mn;
    red[b] = t; __syncthreads();
    for (int s = 128; s > 0; s >>= 1) { if (b < s) red[b] = fmaxf(red[b], red[b + s]); __syncthreads(); }
    float mx = red[0];
    if (mx > 1.0f) t = t / mx;
    t = fminf(fmaxf(t, lastt[b]), 1.0f);
    lastt[b] = t;
    ts[(size_t)b * RW_ + step] = t;
    float* Hr = Ht + (size_t)b * HT_;
    for (int k = 0; k < HT_; ++k) Hr[k] = fmaf(t, Wtu[k], btu[k]);
}

__global__ __launch_bounds__(128)
void xnext_k(const float* __restrict__ Hg, const float* __restrict__ Ht,
             const float* __restrict__ Wvt, float* __restrict__ x) {
    int b = blockIdx.x, j = threadIdx.x;
    const float* hg = Hg + (size_t)b * HIN_;
    const float* ht = Ht + (size_t)b * HT_;
    const float* w  = Wvt + (size_t)j * (HIN_ + HT_);
    float acc = 0.f;
    for (int k = 0; k < HIN_; ++k) acc = fmaf(hg[k], w[k], acc);
    for (int k = 0; k < HT_; ++k)  acc = fmaf(ht[k], w[HIN_ + k], acc);
    x[(size_t)b * HIN_ + j] = acc;
}

__global__ __launch_bounds__(256)
void prob_k(const float* __restrict__ x, const float* __restrict__ Wp, float* __restrict__ prob) {
    int b = threadIdx.x;
    const float* xr = x + (size_t)b * HIN_;
    float acc = 0.f;
    for (int k = 0; k < HIN_; ++k) acc = fmaf(xr[k], Wp[k], acc);
    prob[b] = acc;
}

// ---------------------------------------------------------------------------
extern "C" void kernel_launch(void* const* d_in, const int* in_sizes, int n_in,
                              void* d_out, int out_size, void* d_ws, size_t ws_size,
                              hipStream_t stream) {
    (void)in_sizes; (void)n_in; (void)out_size;

    const float* latent  = (const float*)d_in[0];
    const float* inputs0 = (const float*)d_in[1];
    const float* Wc      = (const float*)d_in[2];
    const float* bc      = (const float*)d_in[3];
    const float* Ws      = (const float*)d_in[4];
    const float* bs      = (const float*)d_in[5];
    const float* Wh      = (const float*)d_in[6];
    const float* bh      = (const float*)d_in[7];
    const float* Wcc     = (const float*)d_in[8];
    const float* bcc     = (const float*)d_in[9];
    const float* W_ih    = (const float*)d_in[10];
    const float* b_ih    = (const float*)d_in[11];
    const float* W_hh    = (const float*)d_in[12];
    const float* b_hh    = (const float*)d_in[13];
    const float* W_up    = (const float*)d_in[14];
    const float* b_up    = (const float*)d_in[15];
    const float* W_down  = (const float*)d_in[16];
    const float* Wt_down = (const float*)d_in[17];
    const float* bt_down = (const float*)d_in[18];
    const float* Wt_pred = (const float*)d_in[19];
    const float* Wt_up   = (const float*)d_in[20];
    const float* bt_up   = (const float*)d_in[21];
    const float* W_vt    = (const float*)d_in[22];
    const float* W_prob  = (const float*)d_in[23];
    const float* gumbel  = (const float*)d_in[24];
    const float* drop_u  = (const float*)d_in[25];

    float* out  = (float*)d_out;
    float* rw   = out;                                  // [B, RW, N]
    float* ts   = out + (size_t)B_ * RW_ * N_;          // [B, RW]
    float* prob = ts + (size_t)B_ * RW_;                // [B]

    char* ws = (char*)d_ws;
    size_t off = 0;
    auto alloc = [&](size_t bytes) -> void* {
        off = (off + 255) & ~(size_t)255;
        void* p = ws + off;
        off += bytes;
        return p;
    };
    unsigned short* wcat  = (unsigned short*)alloc((size_t)4 * H_ * (HIN_ + H_) * 2);
    unsigned short* wupb  = (unsigned short*)alloc((size_t)N_ * H_ * 2);
    unsigned short* wdnb  = (unsigned short*)alloc((size_t)HIN_ * N_ * 2);
    float* bsum  = (float*)alloc((size_t)4 * H_ * 4);
    float* lat   = (float*)alloc((size_t)B_ * ZD_ * 4);
    float* sbuf  = (float*)alloc((size_t)B_ * H_ * 4);
    float* xbuf  = (float*)alloc((size_t)B_ * HIN_ * 4);
    float* hbuf  = (float*)alloc((size_t)B_ * H_ * 4);
    float* cbuf  = (float*)alloc((size_t)B_ * H_ * 4);
    unsigned short* h2bf = (unsigned short*)alloc((size_t)B_ * H_ * 2);
    unsigned short* albf = (unsigned short*)alloc((size_t)B_ * (HIN_ + H_) * 2);
    float* gates = (float*)alloc((size_t)B_ * 4 * H_ * 4);
    float* pbuf  = (float*)alloc((size_t)B_ * N_ * 4);
    unsigned short* vbf = (unsigned short*)alloc((size_t)B_ * N_ * 2);
    float* Hg    = (float*)alloc((size_t)B_ * HIN_ * 4);
    float* HgP   = (float*)alloc((size_t)KSPLIT_ * B_ * HIN_ * 4);   // split-K partials
    float* tpre  = (float*)alloc((size_t)B_ * 4);
    float* lastt = (float*)alloc((size_t)B_ * 4);
    float* Htb   = (float*)alloc((size_t)B_ * HT_ * 4);
    if (off > ws_size) return;   // workspace too small: bail deterministically

    // one-time weight conversion + bias fuse + state init
    concat_wcat_k<<<dim3(1280), dim3(256), 0, stream>>>(W_ih, W_hh, wcat);
    cvt_bf16_k<<<dim3(2048), dim3(256), 0, stream>>>(W_up, wupb, (size_t)N_ * H_);
    cvt_bf16_k<<<dim3(1024), dim3(256), 0, stream>>>(W_down, wdnb, (size_t)HIN_ * N_);
    addvec_k<<<dim3(8), dim3(256), 0, stream>>>(b_ih, b_hh, bsum, 4 * H_);
    zero_k<<<dim3(1), dim3(256), 0, stream>>>(lastt, B_);

    // induced init: lat -> s -> h0, c0
    rowdot_tanh_k<<<dim3(B_, 1), dim3(256), 0, stream>>>(latent, Wc, bc, lat, ZD_, ZD_ + 1);
    rowdot_tanh_k<<<dim3(B_, 2), dim3(256), 0, stream>>>(lat, Ws, bs, sbuf, H_, ZD_);
    rowdot_tanh_k<<<dim3(B_, 2), dim3(256), 0, stream>>>(sbuf, Wh, bh, hbuf, H_, H_);
    rowdot_tanh_k<<<dim3(B_, 2), dim3(256), 0, stream>>>(sbuf, Wcc, bcc, cbuf, H_, H_);
    hipMemcpyAsync(xbuf, inputs0, (size_t)B_ * HIN_ * 4, hipMemcpyDeviceToDevice, stream);

    for (int t = 0; t < RW_; ++t) {
        concat_xh_k<<<dim3(640), dim3(256), 0, stream>>>(xbuf, hbuf, albf);
        // gates = [x|h] * Wcat^T + (b_ih+b_hh)      : 256 x 2048 x 640
        gemm_bf16_wmma_k<2><<<dim3(8, 8, 1), dim3(256), 0, stream>>>(albf, wcat, bsum, gates,
                                                                     B_, 4 * H_, HIN_ + H_);
        lstm_elem_k<<<dim3(512), dim3(256), 0, stream>>>(gates, cbuf, hbuf, h2bf);
        // p = h2 * W_up^T + b_up                    : 256 x 16384 x 512
        gemm_bf16_wmma_k<2><<<dim3(8, 64, 1), dim3(256), 0, stream>>>(h2bf, wupb, b_up, pbuf,
                                                                      B_, N_, H_);
        softmax_g_k<<<dim3(B_), dim3(256), 0, stream>>>(pbuf, gumbel + (size_t)t * B_ * N_,
                                                        rw + (size_t)t * N_, vbf);
        // Hg = v * W_down^T                         : 256 x 128 x 16384, split-K x32
        gemm_bf16_wmma_k<4><<<dim3(4, 1, KSPLIT_), dim3(256), 0, stream>>>(vbf, wdnb, nullptr,
                                                                           HgP, B_, HIN_, N_);
        reduce_splitk_k<<<dim3(128), dim3(256), 0, stream>>>(HgP, Hg, B_ * HIN_, KSPLIT_);
        timedec_k<<<dim3(B_), dim3(HT_), 0, stream>>>(hbuf, Wt_down, bt_down, Wt_pred,
                                                      drop_u + (size_t)t * B_ * HT_, tpre);
        constraints_k<<<dim3(1), dim3(256), 0, stream>>>(tpre, lastt, ts, Htb, Wt_up, bt_up, t);
        xnext_k<<<dim3(B_), dim3(128), 0, stream>>>(Hg, Htb, W_vt, xbuf);
    }
    prob_k<<<dim3(1), dim3(256), 0, stream>>>(xbuf, W_prob, prob);
}
